// SudokuRRN_7730941133188
// MI455X (gfx1250) — matrix operand order repaired and checked by reading, not compiled, not run on hardware
//
#include <hip/hip_runtime.h>

typedef __attribute__((ext_vector_type(16))) __bf16 v16bf;
typedef __attribute__((ext_vector_type(8)))  float  v8f;

#define NN 81
#define HH 128
#define EE 1620
#define BB 32
#define STEPS 16

// ---------------------------------------------------------------------------
// WMMA helpers. B-matrix is pre-packed as [kt][nt][lane][16] bf16 where
// lane<16 holds (K=kt*32+e+(e&8), N=nt*16+lane) and lane>=16 holds
// (K=kt*32+8+e+(e&8), N=nt*16+lane-16) -- the 32x16 bf16 B fragment layout.
// A is read from an LDS row-major activation tile [16][rowStride] per the
// 16-bit 16x32 A layout: lane L -> row L%16, chunks {K0..7,K16..23} (lo lanes)
// or {K8..15,K24..31} (hi lanes).
// ---------------------------------------------------------------------------
template <int KT>
static __device__ __forceinline__ v8f wmma_acc(const __bf16* actIn, int rowStride,
                                               const __bf16* wp, int nt, int lane) {
  v8f c = {};
  const int rr = lane & 15;
  const int hi = (lane & 16) ? 8 : 0;
#pragma unroll
  for (int kt = 0; kt < KT; ++kt) {
    const __bf16* ap = actIn + rr * rowStride + kt * 32 + hi;
    union { v16bf v; uint4 q[2]; } A, B;
    A.q[0] = *(const uint4*)(ap);
    A.q[1] = *(const uint4*)(ap + 16);
    const __bf16* bp = wp + (size_t)((kt * 8 + nt) * 32 + lane) * 16;
    B.q[0] = *(const uint4*)(bp);
    B.q[1] = *(const uint4*)(bp + 8);
    c = __builtin_amdgcn_wmma_f32_16x16x32_bf16(false, A.v, false, B.v,
                                                (short)0, c, false, false);
  }
  return c;
}

// One full 16xK -> 16x128 layer: 8 N-tiles, bias + optional relu, bf16 out to LDS.
template <int KT>
static __device__ __forceinline__ void mlp_layer(const __bf16* actIn, int rowStride,
                                                 const __bf16* wp, const float* bias,
                                                 __bf16* actOut, int lane, bool relu) {
  const int col0  = lane & 15;
  const int rbase = (lane & 16) ? 8 : 0;
  for (int nt = 0; nt < 8; ++nt) {
    v8f c = wmma_acc<KT>(actIn, rowStride, wp, nt, lane);
    const int col = nt * 16 + col0;
    const float bc = bias[col];
#pragma unroll
    for (int v = 0; v < 8; ++v) {
      float xv = c[v] + bc;
      if (relu) xv = fmaxf(xv, 0.0f);
      actOut[(rbase + v) * HH + col] = (__bf16)xv;
    }
  }
}

// ---------------------------------------------------------------------------
// Weight pack: W[K][128] f32 row-major -> bf16 B-fragment tiles.
// dst[tid] with tid = kt*4096 + nt*512 + lane*16 + e.
// ---------------------------------------------------------------------------
__global__ void pack_kernel(const float* __restrict__ W, __bf16* __restrict__ dst, int K) {
  int tid = blockIdx.x * blockDim.x + threadIdx.x;
  if (tid >= K * 128) return;
  int e    = tid & 15;
  int lane = (tid >> 4) & 31;
  int nt   = (tid >> 9) & 7;
  int kt   = tid >> 12;
  int kidx = kt * 32 + ((e >> 3) << 4) + ((lane >> 4) << 3) + (e & 7);
  int col  = nt * 16 + (lane & 15);
  dst[tid] = (__bf16)(W[kidx * 128 + col]);
}

// ---------------------------------------------------------------------------
// Embed: x@Wi + bi + pos  -> hf (f32), hb/xeb (bf16); zero agg.
// ---------------------------------------------------------------------------
__global__ void embed_kernel(const float* __restrict__ x, const float* __restrict__ Wi,
                             const float* __restrict__ bi, const float* __restrict__ pos,
                             float* __restrict__ hf, __bf16* __restrict__ hb,
                             __bf16* __restrict__ xeb, float* __restrict__ agg) {
  int tid = blockIdx.x * blockDim.x + threadIdx.x;
  if (tid >= BB * NN * HH) return;
  int c = tid & (HH - 1);
  int n = (tid >> 7) % NN;
  int b = tid / (NN * HH);
  const float* xr = x + ((size_t)b * NN + n) * 10;
  float s = bi[c] + pos[n * HH + c];
#pragma unroll
  for (int k = 0; k < 10; ++k) s += xr[k] * Wi[k * HH + c];
  hf[tid]  = s;
  hb[tid]  = (__bf16)s;
  xeb[tid] = (__bf16)s;
  agg[tid] = 0.0f;
}

// ---------------------------------------------------------------------------
// Message kernel: one wave per 16-edge tile per batch. Gathers concat(h[src],
// h[dst]) into LDS, runs the 4-layer MLP with WMMA, scatter-adds into agg.
// ---------------------------------------------------------------------------
__global__ __launch_bounds__(32) void msg_kernel(
    const __bf16* __restrict__ hb, float* __restrict__ agg,
    const int* __restrict__ esrc, const int* __restrict__ edst,
    const __bf16* __restrict__ wp1, const __bf16* __restrict__ wp2,
    const __bf16* __restrict__ wp3, const __bf16* __restrict__ wp4,
    const float* __restrict__ b1, const float* __restrict__ b2,
    const float* __restrict__ b3, const float* __restrict__ b4) {
  __shared__ __align__(16) __bf16 actA[16 * 256];
  __shared__ __align__(16) __bf16 actB[16 * HH];
  __shared__ int dstIdx[16];

  const int lane = threadIdx.x;
  const int tile = blockIdx.x;
  const int b    = blockIdx.y;
  const int base = tile * 16;

  {  // gather concat input tile [16][256] bf16
    int r = lane & 15;
    int e = base + r; if (e > EE - 1) e = EE - 1;
    int node = (lane < 16) ? esrc[e] : edst[e];
    if (lane >= 16) dstIdx[r] = node;
    const __bf16* s = hb + ((size_t)b * NN + node) * HH;
    __bf16* d = actA + r * 256 + ((lane < 16) ? 0 : HH);
#pragma unroll
    for (int i = 0; i < 16; ++i) ((uint4*)d)[i] = ((const uint4*)s)[i];
  }
  __syncthreads();

  mlp_layer<8>(actA, 256, wp1, b1, actB, lane, true);
  __syncthreads();
  mlp_layer<4>(actB, HH, wp2, b2, actA, lane, true);   // reuse actA as [16][128]
  __syncthreads();
  mlp_layer<4>(actA, HH, wp3, b3, actB, lane, true);
  __syncthreads();

  {  // layer 4: no relu, scatter-add per dst
    const int col0  = lane & 15;
    const int rbase = (lane & 16) ? 8 : 0;
    for (int nt = 0; nt < 8; ++nt) {
      v8f c = wmma_acc<4>(actB, HH, wp4, nt, lane);
      const int col = nt * 16 + col0;
      const float bc = b4[col];
#pragma unroll
      for (int v = 0; v < 8; ++v) {
        int row = rbase + v;
        if (base + row < EE)
          atomicAdd(agg + ((size_t)b * NN + dstIdx[row]) * HH + col, c[v] + bc);
      }
    }
  }
}

// ---------------------------------------------------------------------------
// Node kernel: one wave per 16-node tile per batch. concat(h, x_embed, agg)
// -> 4-layer MLP (WMMA) -> residual + LayerNorm -> h (f32 + bf16 shadow).
// Also zeroes agg for the next step after consuming it.
// ---------------------------------------------------------------------------
__global__ __launch_bounds__(32) void node_kernel(
    __bf16* __restrict__ hb, const __bf16* __restrict__ xeb,
    float* __restrict__ agg, float* __restrict__ hf,
    const __bf16* __restrict__ wp1, const __bf16* __restrict__ wp2,
    const __bf16* __restrict__ wp3, const __bf16* __restrict__ wp4,
    const float* __restrict__ b1, const float* __restrict__ b2,
    const float* __restrict__ b3, const float* __restrict__ b4,
    const float* __restrict__ gamma, const float* __restrict__ beta) {
  __shared__ __align__(16) __bf16 actA[16 * 384];
  __shared__ __align__(16) __bf16 actB[16 * HH];
  __shared__ __align__(16) float  outF[16 * HH];

  const int lane = threadIdx.x;
  const int tile = blockIdx.x;
  const int b    = blockIdx.y;
  const int r    = lane & 15;
  int n = tile * 16 + r; if (n > NN - 1) n = NN - 1;
  const size_t rowOff = ((size_t)b * NN + n) * HH;

  {  // gather [16][384]: cols 0..127 = h, 128..255 = x_embed, 256..383 = agg
    const __bf16* s = ((lane < 16) ? hb : xeb) + rowOff;
    __bf16* d = actA + r * 384 + ((lane < 16) ? 0 : HH);
#pragma unroll
    for (int i = 0; i < 16; ++i) ((uint4*)d)[i] = ((const uint4*)s)[i];
    const int cbase = (lane >> 4) * 64;
    for (int i = 0; i < 64; ++i)
      actA[r * 384 + 256 + cbase + i] = (__bf16)agg[rowOff + cbase + i];
  }
  __syncthreads();

  {  // zero agg for next step (valid rows only, each owned by exactly one block)
    int nv = tile * 16 + r;
    if (nv < NN) {
      const int cbase = (lane >> 4) * 64;
      float* ap = agg + ((size_t)b * NN + nv) * HH + cbase;
      for (int i = 0; i < 64; ++i) ap[i] = 0.0f;
    }
  }

  mlp_layer<12>(actA, 384, wp1, b1, actB, lane, true);
  __syncthreads();
  mlp_layer<4>(actB, HH, wp2, b2, actA, lane, true);   // reuse actA as [16][128]
  __syncthreads();
  mlp_layer<4>(actA, HH, wp3, b3, actB, lane, true);
  __syncthreads();

  {  // layer 4 -> f32 LDS staging
    const int col0  = lane & 15;
    const int rbase = (lane & 16) ? 8 : 0;
    for (int nt = 0; nt < 8; ++nt) {
      v8f c = wmma_acc<4>(actB, HH, wp4, nt, lane);
      const int col = nt * 16 + col0;
      const float bc = b4[col];
#pragma unroll
      for (int v = 0; v < 8; ++v) outF[(rbase + v) * HH + col] = c[v] + bc;
    }
  }
  __syncthreads();

  if (lane < 16) {  // residual + LayerNorm, one lane per row
    int nv = tile * 16 + lane;
    if (nv < NN) {
      const size_t ro = ((size_t)b * NN + nv) * HH;
      float mu = 0.0f, sq = 0.0f;
      for (int c2 = 0; c2 < HH; ++c2) {
        float v = hf[ro + c2] + outF[lane * HH + c2];
        mu += v; sq += v * v;
      }
      mu *= (1.0f / HH);
      float var = sq * (1.0f / HH) - mu * mu;
      float inv = rsqrtf(var + 1e-5f);
      for (int c2 = 0; c2 < HH; ++c2) {
        float v = hf[ro + c2] + outF[lane * HH + c2];
        float o = (v - mu) * inv * gamma[c2] + beta[c2];
        hf[ro + c2] = o;
        hb[ro + c2] = (__bf16)o;
      }
    }
  }
}

// ---------------------------------------------------------------------------
// Output head: h @ Wo + bo, tiny (32*81*9).
// ---------------------------------------------------------------------------
__global__ void out_kernel(const float* __restrict__ hf, const float* __restrict__ Wo,
                           const float* __restrict__ bo, float* __restrict__ out) {
  int tid = blockIdx.x * blockDim.x + threadIdx.x;
  if (tid >= BB * NN * 9) return;
  int j  = tid % 9;
  int bn = tid / 9;
  const float* hr = hf + (size_t)bn * HH;
  float s = bo[j];
  for (int c = 0; c < HH; ++c) s += hr[c] * Wo[c * 9 + j];
  out[tid] = s;
}

// ---------------------------------------------------------------------------
extern "C" void kernel_launch(void* const* d_in, const int* in_sizes, int n_in,
                              void* d_out, int out_size, void* d_ws, size_t ws_size,
                              hipStream_t stream) {
  const float* x    = (const float*)d_in[0];
  const int*   ei   = (const int*)d_in[1];
  const float* Wi   = (const float*)d_in[2];
  const float* bi   = (const float*)d_in[3];
  const float* pos  = (const float*)d_in[4];
  const float* mW1  = (const float*)d_in[5];  const float* mb1 = (const float*)d_in[6];
  const float* mW2  = (const float*)d_in[7];  const float* mb2 = (const float*)d_in[8];
  const float* mW3  = (const float*)d_in[9];  const float* mb3 = (const float*)d_in[10];
  const float* mW4  = (const float*)d_in[11]; const float* mb4 = (const float*)d_in[12];
  const float* nW1  = (const float*)d_in[13]; const float* nb1 = (const float*)d_in[14];
  const float* nW2  = (const float*)d_in[15]; const float* nb2 = (const float*)d_in[16];
  const float* nW3  = (const float*)d_in[17]; const float* nb3 = (const float*)d_in[18];
  const float* nW4  = (const float*)d_in[19]; const float* nb4 = (const float*)d_in[20];
  const float* gam  = (const float*)d_in[21]; const float* bet = (const float*)d_in[22];
  const float* Wo   = (const float*)d_in[23]; const float* bo  = (const float*)d_in[24];

  const int* esrc = ei;
  const int* edst = ei + EE;

  char* p = (char*)d_ws;
  float*  hf  = (float*)p;  p += (size_t)BB * NN * HH * 4;
  __bf16* hb  = (__bf16*)p; p += (size_t)BB * NN * HH * 2;
  __bf16* xeb = (__bf16*)p; p += (size_t)BB * NN * HH * 2;
  float*  agg = (float*)p;  p += (size_t)BB * NN * HH * 4;
  __bf16* wm1 = (__bf16*)p; p += 256 * 128 * 2;
  __bf16* wm2 = (__bf16*)p; p += 128 * 128 * 2;
  __bf16* wm3 = (__bf16*)p; p += 128 * 128 * 2;
  __bf16* wm4 = (__bf16*)p; p += 128 * 128 * 2;
  __bf16* wn1 = (__bf16*)p; p += 384 * 128 * 2;
  __bf16* wn2 = (__bf16*)p; p += 128 * 128 * 2;
  __bf16* wn3 = (__bf16*)p; p += 128 * 128 * 2;
  __bf16* wn4 = (__bf16*)p; p += 128 * 128 * 2;

  auto pack = [&](const float* W, __bf16* dst, int K) {
    int n = K * 128;
    pack_kernel<<<(n + 255) / 256, 256, 0, stream>>>(W, dst, K);
  };
  pack(mW1, wm1, 256); pack(mW2, wm2, 128); pack(mW3, wm3, 128); pack(mW4, wm4, 128);
  pack(nW1, wn1, 384); pack(nW2, wn2, 128); pack(nW3, wn3, 128); pack(nW4, wn4, 128);

  {
    int n = BB * NN * HH;
    embed_kernel<<<(n + 255) / 256, 256, 0, stream>>>(x, Wi, bi, pos, hf, hb, xeb, agg);
  }

  const int etiles = (EE + 15) / 16;  // 102
  const int ntiles = (NN + 15) / 16;  // 6
  for (int s = 0; s < STEPS; ++s) {
    msg_kernel<<<dim3(etiles, BB), 32, 0, stream>>>(hb, agg, esrc, edst,
                                                    wm1, wm2, wm3, wm4,
                                                    mb1, mb2, mb3, mb4);
    node_kernel<<<dim3(ntiles, BB), 32, 0, stream>>>(hb, xeb, agg, hf,
                                                     wn1, wn2, wn3, wn4,
                                                     nb1, nb2, nb3, nb4, gam, bet);
  }

  {
    int n = BB * NN * 9;
    out_kernel<<<(n + 255) / 256, 256, 0, stream>>>(hf, Wo, bo, (float*)d_out);
  }
}
